// ConvFourier_33148557590581
// MI455X (gfx1250) — matrix operand order repaired and checked by reading, not compiled, not run on hardware
//
#include <hip/hip_runtime.h>

// ConvFourier on MI455X (gfx1250, wave32, WMMA).
//
// out[b,f,oh,ow] = sum_sl sum_k [ cos(sl*p)*Wc + sin(sl*p)*Ws ] + bias[f]/2
//   p = im2col patch (C=32, 3x3), K = 288 = 9 chunks of 32 (reordered as kh,kw,c)
//   16 GEMMs  M=254016  K=288  N=64  -> v_wmma_f32_16x16x32_f16, f32 accum.
// Trig reduced 8x via angle-addition recurrence; the running (cos,sin) state is
// kept in packed f16 so the WMMA A fragments ARE the state registers.
// Each wave owns TWO 16-row A tiles so every B fragment load feeds 2 WMMAs
// (halves L0 B-traffic per matrix op).  sl=0 folded into the bias term.

typedef __attribute__((ext_vector_type(16))) _Float16 v16h;
typedef __attribute__((ext_vector_type(8)))  float    v8f;

#define BATCH 16
#define CIN   32
#define HW    128
#define FOUT  64
#define SLN   8
#define OHW   126           // 128 - 3 + 1
#define CKK   288
#define NFRAG_HALFS (8 * 2 * 9 * 4 * 32 * 16)   // sl,trig,kc,ntile,lane,16 halfs = 294912

// ---------------------------------------------------------------------------
// Repack weights fp32 -> f16 B-fragments for v_wmma_f32_16x16x32_f16.
// B is 32(K) x 16(N): lanes 0-15 hold n=lane, K=0..15 (element e -> K=e);
// lanes 16-31 hold n=lane-16, K=16..31 (element e -> K=16+e).  K == channel,
// chunk kc == spatial tap (kh,kw).
// ---------------------------------------------------------------------------
__global__ __launch_bounds__(256) void repack_weights(const float* __restrict__ w,
                                                      _Float16* __restrict__ wh) {
    int idx = blockIdx.x * 256 + threadIdx.x;
    if (idx >= NFRAG_HALFS) return;
    int e     = idx & 15;
    int laneb = (idx >> 4) & 31;
    int nt    = (idx >> 9) & 3;
    int kc    = (idx >> 11) % 9;
    int rest  = (idx >> 11) / 9;        // trig + 2*sl
    int trig  = rest & 1;
    int sl    = rest >> 1;
    int c     = (laneb < 16) ? e : (16 + e);
    int f     = nt * 16 + (laneb & 15);
    int kh    = kc / 3, kw = kc - kh * 3;
    // weight layout: [F][SL][2][C][3][3]
    float v = w[((((f * SLN + sl) * 2 + trig) * CIN + c) * 3 + kh) * 3 + kw];
    wh[idx] = (_Float16)v;
}

// bias2[f] = 0.5*bias[f] + sum_k Wc[sl=0,f,k]   (cos(0)=1, sin(0)=0 term)
__global__ void prep_bias(const float* __restrict__ w, const float* __restrict__ bias,
                          float* __restrict__ bias2) {
    int f = threadIdx.x;                 // 64 threads
    const float* p = w + (size_t)f * (SLN * 2 * CKK);   // sl=0, trig=0 block: 288 contiguous
    float s = 0.f;
    for (int t = 0; t < CKK; ++t) s += p[t];
    bias2[f] = 0.5f * bias[f] + s;
}

// ---------------------------------------------------------------------------
// Main kernel: one workgroup (128 thr = 4 waves) per (b, oh) output row.
// Each wave owns a 32(M) x 64(N) tile: 2 A-subtiles x 4 N-tiles = 8 f32 accs,
// so each B fragment load is reused by two WMMAs.
// ---------------------------------------------------------------------------
__global__ __launch_bounds__(128) void fourier_main(const float* __restrict__ x,
                                                    const _Float16* __restrict__ wh,
                                                    const float* __restrict__ bias2,
                                                    float* __restrict__ out) {
    // Padded col dim (132): avoids lane<->lane+16 LDS bank conflicts and keeps
    // the (masked) rows 126/127 reads in-range.  32*3*132*4 = 50688 B.
    __shared__ float xs[CIN][3][132];

    int b  = blockIdx.x / OHW;
    int oh = blockIdx.x - b * OHW;
    int tid = threadIdx.x;

    // Stage x[b, 0..31, oh..oh+2, 0..127] into LDS (coalesced float4 loads).
    const float* xb = x + (size_t)b * CIN * HW * HW;
    for (int i = tid; i < CIN * 3 * 32; i += 128) {      // 3072 float4s, 24/thread
        int c   = i / 96;
        int rem = i - c * 96;
        int r   = rem >> 5;
        int c4  = (rem & 31) << 2;
        const float4 v = *(const float4*)(xb + ((size_t)c * HW + (oh + r)) * HW + c4);
        xs[c][r][c4 + 0] = v.x;
        xs[c][r][c4 + 1] = v.y;
        xs[c][r][c4 + 2] = v.z;
        xs[c][r][c4 + 3] = v.w;
    }
    __syncthreads();

    int lane  = tid & 31;
    int wave  = tid >> 5;                // 0..3
    int m0    = wave << 5;               // 32 output columns per wave
    int cbase = (lane < 16) ? 0 : 8;     // A 16x32 f16 K-layout split

    v8f acc[2][4] = {};                  // [m-subtile][n-tile] 16x16 f32 C/D tiles

    #pragma unroll 1
    for (int kc = 0; kc < 9; ++kc) {
        int kh = kc / 3, kw = kc - kh * 3;

        // Trig of the phase, once per patch element, stored straight into
        // packed f16 in the WMMA A-fragment element order (two M-subtiles).
        v16h c1h[2], s1h[2], cc[2], ss[2];
        #pragma unroll
        for (int h = 0; h < 2; ++h) {
            int mrow = m0 + h * 16 + (lane & 15);   // A-matrix row for this lane
            #pragma unroll
            for (int j = 0; j < 16; ++j) {
                int c    = ((lane >> 4) << 4) + j;
                float p  = xs[c][kh][mrow + kw];
                c1h[h][j] = (_Float16)__cosf(p);
                s1h[h][j] = (_Float16)__sinf(p);
            }
            cc[h] = c1h[h];              // cos(sl*p), sl=1
            ss[h] = s1h[h];              // sin(sl*p)
        }

        #pragma unroll
        for (int sl = 1; sl < 8; ++sl) {
            // fragment base: ((((sl*2+trig)*9 + kc)*4 + nt)*32 + lane)*16 halfs
            const _Float16* fc = wh + ((size_t)(((sl * 2 + 0) * 9 + kc) * 4) * 32 + lane) * 16;
            const _Float16* fs = fc + 9 * 4 * 32 * 16;   // +1 trig plane
            #pragma unroll
            for (int nt = 0; nt < 4; ++nt) {
                v16h bc = *(const v16h*)(fc + nt * 512);
                acc[0][nt] = __builtin_amdgcn_wmma_f32_16x16x32_f16(
                    false, cc[0], false, bc, (short)0, acc[0][nt], false, false);
                acc[1][nt] = __builtin_amdgcn_wmma_f32_16x16x32_f16(
                    false, cc[1], false, bc, (short)0, acc[1][nt], false, false);
                v16h bs = *(const v16h*)(fs + nt * 512);
                acc[0][nt] = __builtin_amdgcn_wmma_f32_16x16x32_f16(
                    false, ss[0], false, bs, (short)0, acc[0][nt], false, false);
                acc[1][nt] = __builtin_amdgcn_wmma_f32_16x16x32_f16(
                    false, ss[1], false, bs, (short)0, acc[1][nt], false, false);
            }
            // angle addition in packed f16 (v_pk_fma_f16): the next A fragments
            // are produced directly in the state registers.
            if (sl < 7) {
                #pragma unroll
                for (int h = 0; h < 2; ++h) {
                    v16h nc = cc[h] * c1h[h] - ss[h] * s1h[h];
                    ss[h]   = ss[h] * c1h[h] + cc[h] * s1h[h];
                    cc[h]   = nc;
                }
            }
        }
    }

    // Epilogue.  C/D layout: VGPR r -> M = r (+8 for lanes>=16), lane%16 -> N.
    int nrow = lane & 15;
    int mofs = (lane >> 4) << 3;
    float* ob = out + ((size_t)b * FOUT * OHW + oh) * OHW;
    #pragma unroll
    for (int h = 0; h < 2; ++h) {
        #pragma unroll
        for (int nt = 0; nt < 4; ++nt) {
            int f    = nt * 16 + nrow;
            float bb = bias2[f];
            #pragma unroll
            for (int r = 0; r < 8; ++r) {
                int m = m0 + h * 16 + mofs + r;
                if (m < OHW) ob[(size_t)f * OHW * OHW + m] = acc[h][nt][r] + bb;
            }
        }
    }
}

// ---------------------------------------------------------------------------
extern "C" void kernel_launch(void* const* d_in, const int* in_sizes, int n_in,
                              void* d_out, int out_size, void* d_ws, size_t ws_size,
                              hipStream_t stream) {
    const float* x    = (const float*)d_in[0];
    const float* w    = (const float*)d_in[1];
    const float* bias = (const float*)d_in[2];
    float* out        = (float*)d_out;

    _Float16* wh  = (_Float16*)d_ws;
    float* bias2  = (float*)((char*)d_ws + (size_t)NFRAG_HALFS * sizeof(_Float16));

    repack_weights<<<(NFRAG_HALFS + 255) / 256, 256, 0, stream>>>(w, wh);
    prep_bias<<<1, 64, 0, stream>>>(w, bias, bias2);
    fourier_main<<<BATCH * OHW, 128, 0, stream>>>(x, wh, bias2, out);
}
